// EnhancedFreqFeature_32315333935259
// MI455X (gfx1250) — matrix operand order repaired and checked by reading, not compiled, not run on hardware
//
#include <hip/hip_runtime.h>
#include <hip/hip_bf16.h>
#include <math.h>

typedef __attribute__((ext_vector_type(16))) _Float16 v16h;
typedef __attribute__((ext_vector_type(8)))  float    v8f;

union U16h { v16h v; _Float16 h[16]; };
union U8f  { v8f  v; float    f[8];  };

#define WMMA_F16(A, B, C) __builtin_amdgcn_wmma_f32_16x16x32_f16(false, (A), false, (B), (short)0, (C), false, false)

// ---------------------------------------------------------------------------
// Init kernels: pack DFT twiddles and conv weights into WMMA B-matrix lane
// order, split into hi/lo f16 for the 3-pass split-precision GEMM.
// B packing used here (contiguous k, 32x16 per K-step): lane n<16 -> k_local = e (0..15),
// lane n>=16 -> k_local = 16+e ; col = n & 15.
// ---------------------------------------------------------------------------
__global__ void pack_twiddle_kernel(_Float16* __restrict__ hi,
                                    _Float16* __restrict__ lo,
                                    int NFFT, int BINS, int NTH, int KSTEPS) {
    int tid = blockIdx.x * 256 + threadIdx.x;
    int total = 2 * NTH * KSTEPS * 512;
    if (tid >= total) return;
    int e    = tid & 15;
    int lane = (tid >> 4) & 31;
    int ts   = tid >> 9;                 // t*KSTEPS + s
    int s    = ts % KSTEPS;
    int t    = ts / KSTEPS;
    int k    = s * 32 + ((lane >> 4) << 4) + e;     // 0..NFFT-1
    int col  = (t % NTH) * 16 + (lane & 15);
    bool isim = (t >= NTH);
    float v = 0.0f;
    if (col < BINS) {
        int prod = (col * k) % NFFT;
        float ang = -6.283185307179586f * (float)prod / (float)NFFT;
        v = isim ? sinf(ang) : cosf(ang);   // re: cos(2pi ck/N), im: -sin(2pi ck/N)
    }
    _Float16 h = (_Float16)v;
    hi[tid] = h;
    lo[tid] = (_Float16)(v - (float)h);
}

__global__ void pack_convw_kernel(const float* __restrict__ w,  // [64][32][3]
                                  _Float16* __restrict__ hi,
                                  _Float16* __restrict__ lo) {
    int tid = blockIdx.x * 256 + threadIdx.x;   // total 12*512 = 6144
    if (tid >= 6144) return;
    int e    = tid & 15;
    int lane = (tid >> 4) & 31;
    int nts  = tid >> 9;                 // nt*3 + s
    int s    = nts % 3;
    int nt   = nts / 3;
    int k    = s * 32 + ((lane >> 4) << 4) + e;     // 0..95 = c*3+dk
    int o    = nt * 16 + (lane & 15);
    float v = w[o * 96 + k];
    _Float16 h = (_Float16)v;
    hi[tid] = h;
    lo[tid] = (_Float16)(v - (float)h);
}

// ---------------------------------------------------------------------------
// DFT-as-GEMM. Each wave owns a 16-row M tile (rows = b*NCH+ch), K = n_fft.
// A layout used here (16x32, contiguous k): lane<16 row=lane, k = 0..15;
// lane>=16 row=lane-16, k = 16..31.
// N space: [re bins | pad][im bins | pad], each half NTH 16-col tiles; a re
// tile t and its im tile t+NTH carry matching (row, bin) in the same
// lane/element, so each pair is computed together and retired immediately
// (mag/phase/mag^2 lane-local) -> only 2 live accumulator tiles (16 VGPRs).
// ---------------------------------------------------------------------------
template <int NCH, int KSTEPS, int BINS, int NTH, int DO_BAND>
__global__ __launch_bounds__(256) void dft_kernel(
        const float* __restrict__ x,          // [B,32,1024]
        const _Float16* __restrict__ twh,
        const _Float16* __restrict__ twl,
        float* __restrict__ feat,             // [B,32,BINS]
        float* __restrict__ bandfeat) {       // [B,160] (DO_BAND only)
    __shared__ float smag2[DO_BAND ? 8 * 16 * 65 : 1];

    const int lane = threadIdx.x & 31;
    const int wid  = threadIdx.x >> 5;
    const int tile = blockIdx.x * 8 + wid;
    const int m0   = tile * 16;
    const int rl   = lane & 15;
    const int hf   = lane >> 4;

    // ---- load + split A (whole K strip kept in registers) ----
    const int rowA = m0 + rl;
    const int bA   = rowA / NCH;
    const int chA  = rowA % NCH;
    const float* xrow = x + ((size_t)bA * 32 + chA) * 1024;
    __builtin_prefetch(xrow, 0, 0);

    U16h ah[KSTEPS], al[KSTEPS];
#pragma unroll
    for (int s = 0; s < KSTEPS; ++s) {
#pragma unroll
        for (int e = 0; e < 16; ++e) {
            int k = s * 32 + hf * 16 + e;
            float v = xrow[k];
            _Float16 h = (_Float16)v;
            ah[s].h[e] = h;
            al[s].h[e] = (_Float16)(v - (float)h);
        }
    }

    const v16h* bh16 = reinterpret_cast<const v16h*>(twh);
    const v16h* bl16 = reinterpret_cast<const v16h*>(twl);

    // ---- process re/im tile pairs, retire each pair immediately ----
#pragma unroll
    for (int t = 0; t < NTH; ++t) {
        U8f accR, accI;
#pragma unroll
        for (int i = 0; i < 8; ++i) { accR.f[i] = 0.0f; accI.f[i] = 0.0f; }

#pragma unroll
        for (int s = 0; s < KSTEPS; ++s) {
            U16h bhR, blR, bhI, blI;
            int idxR = (t * KSTEPS + s) * 32 + lane;
            int idxI = ((t + NTH) * KSTEPS + s) * 32 + lane;
            bhR.v = bh16[idxR];
            blR.v = bl16[idxR];
            bhI.v = bh16[idxI];
            blI.v = bl16[idxI];
            accR.v = WMMA_F16(ah[s].v, bhR.v, accR.v);
            accI.v = WMMA_F16(ah[s].v, bhI.v, accI.v);
            accR.v = WMMA_F16(ah[s].v, blR.v, accR.v);
            accI.v = WMMA_F16(ah[s].v, blI.v, accI.v);
            accR.v = WMMA_F16(al[s].v, bhR.v, accR.v);
            accI.v = WMMA_F16(al[s].v, bhI.v, accI.v);
        }

        // epilogue for this bin group: mag / phase / mag^2
        int f = t * 16 + rl;
#pragma unroll
        for (int e = 0; e < 8; ++e) {
            float re = accR.f[e];
            float im = accI.f[e];
            int m    = e + hf * 8;
            int row  = m0 + m;
            int b    = row / NCH;
            int ch   = row % NCH;
            if (f < BINS) {
                float m2 = re * re + im * im;
                if (ch < 16) {
                    feat[((size_t)b * 32 + ch) * BINS + f]      = sqrtf(m2);
                    feat[((size_t)b * 32 + 16 + ch) * BINS + f] = atan2f(im, re);
                }
                if (DO_BAND) smag2[(wid * 16 + m) * 65 + f] = m2;
            }
        }
    }

    if (DO_BAND) {
        __syncthreads();
        if (lane < 16) {
            const int flo[5] = {1, 4, 8, 13, 30};
            const int fhi[5] = {4, 8, 13, 30, 45};
            int row = m0 + lane;
            int b   = row / NCH;
            int ch  = row % NCH;
            const float* slab = &smag2[(wid * 16 + lane) * 65];
            for (int bd = 0; bd < 5; ++bd) {
                float s = 0.0f;
                for (int f = flo[bd]; f <= fhi[bd]; ++f) s += slab[f];
                bandfeat[(size_t)b * 160 + bd * 32 + ch] =
                    s / (float)(fhi[bd] - flo[bd] + 1);
            }
        }
    }
}

// ---------------------------------------------------------------------------
// Branch head: im2col conv1d(32->64,k3,pad1) as WMMA GEMM (K=96), fused
// bias+BN+GELU+mean epilogue, then 64->SD linear. One wave per batch row.
// ---------------------------------------------------------------------------
template <int NF, int MT, int SD>
__global__ __launch_bounds__(32) void branch_kernel(
        const float* __restrict__ feat,       // [B,32,NF]
        const _Float16* __restrict__ wh,
        const _Float16* __restrict__ wl,
        const float* __restrict__ conv_b,
        const float* __restrict__ bn_g, const float* __restrict__ bn_b,
        const float* __restrict__ bn_m, const float* __restrict__ bn_v,
        const float* __restrict__ lin_w,      // [SD,64]
        const float* __restrict__ lin_b,      // [SD]
        float* __restrict__ comb, int comb_off) {
    __shared__ float sfeat[32 * NF];
    __shared__ float sms[4 * 32];
    __shared__ float smean[64];

    const int lane = threadIdx.x;
    const int b    = blockIdx.x;
    const int rl   = lane & 15;
    const int hf   = lane >> 4;

    const float* fb = feat + (size_t)b * 32 * NF;
    for (int i = lane; i < 32 * NF; i += 32) sfeat[i] = fb[i];
    __syncthreads();

    const v16h* wh16 = reinterpret_cast<const v16h*>(wh);
    const v16h* wl16 = reinterpret_cast<const v16h*>(wl);

    float msum[4] = {0.0f, 0.0f, 0.0f, 0.0f};

    for (int mt = 0; mt < MT; ++mt) {
        // build im2col A tile (rows = positions f, K = c*3+dk)
        U16h aH[3], aL[3];
        int fpos = mt * 16 + rl;
#pragma unroll
        for (int s = 0; s < 3; ++s) {
#pragma unroll
            for (int e = 0; e < 16; ++e) {
                int k   = s * 32 + hf * 16 + e;   // 0..95
                int c   = k / 3;
                int dk  = k - c * 3;
                int src = fpos - 1 + dk;
                float v = (src >= 0 && src < NF) ? sfeat[c * NF + src] : 0.0f;
                _Float16 h = (_Float16)v;
                aH[s].h[e] = h;
                aL[s].h[e] = (_Float16)(v - (float)h);
            }
        }
#pragma unroll
        for (int nt = 0; nt < 4; ++nt) {
            U8f acc;
#pragma unroll
            for (int i = 0; i < 8; ++i) acc.f[i] = 0.0f;
#pragma unroll
            for (int s = 0; s < 3; ++s) {
                U16h bh, bl;
                int idx = (nt * 3 + s) * 32 + lane;
                bh.v = wh16[idx];
                bl.v = wl16[idx];
                acc.v = WMMA_F16(aH[s].v, bh.v, acc.v);
                acc.v = WMMA_F16(aH[s].v, bl.v, acc.v);
                acc.v = WMMA_F16(aL[s].v, bh.v, acc.v);
            }
            int o = nt * 16 + rl;
            float scl = bn_g[o] * rsqrtf(bn_v[o] + 1e-5f);
            float cb  = conv_b[o];
            float bmu = bn_m[o];
            float bbt = bn_b[o];
#pragma unroll
            for (int e = 0; e < 8; ++e) {
                int ff = mt * 16 + e + hf * 8;
                if (ff < NF) {
                    float y = acc.f[e] + cb;
                    y = (y - bmu) * scl + bbt;
                    msum[nt] += 0.5f * y * (1.0f + erff(y * 0.70710678118654752f));
                }
            }
        }
    }

    for (int nt = 0; nt < 4; ++nt) sms[nt * 32 + lane] = msum[nt];
    __syncthreads();
    for (int o = lane; o < 64; o += 32) {
        int nt = o >> 4, j = o & 15;
        smean[o] = (sms[nt * 32 + j] + sms[nt * 32 + 16 + j]) * (1.0f / (float)NF);
    }
    __syncthreads();
    for (int i = lane; i < SD; i += 32) {
        float d = lin_b[i];
        for (int o = 0; o < 64; ++o) d += smean[o] * lin_w[i * 64 + o];
        comb[(size_t)b * 128 + comb_off + i] = d;
    }
}

// ---------------------------------------------------------------------------
// Band MLP (160->128) + LN + GELU, add combined, final LN. ~84 MFLOP total.
// ---------------------------------------------------------------------------
__global__ __launch_bounds__(128) void band_final_kernel(
        const float* __restrict__ bandfeat,   // [B,160]
        const float* __restrict__ comb,       // [B,128]
        const float* __restrict__ band_w,     // [128,160]
        const float* __restrict__ band_b,
        const float* __restrict__ ln_g, const float* __restrict__ ln_b,
        const float* __restrict__ fn_g, const float* __restrict__ fn_b,
        float* __restrict__ out) {
    __shared__ float sv[128];
    const int j = threadIdx.x;
    const int b = blockIdx.x;

    const float* bf = bandfeat + (size_t)b * 160;
    const float* wr = band_w + (size_t)j * 160;
    float d = band_b[j];
    for (int k = 0; k < 160; ++k) d += bf[k] * wr[k];

    sv[j] = d;
    __syncthreads();
    float mu = 0.0f;
    for (int i = 0; i < 128; ++i) mu += sv[i];
    mu *= (1.0f / 128.0f);
    float var = 0.0f;
    for (int i = 0; i < 128; ++i) { float t = sv[i] - mu; var += t * t; }
    var *= (1.0f / 128.0f);
    float yh = (d - mu) * rsqrtf(var + 1e-5f) * ln_g[j] + ln_b[j];
    float g  = 0.5f * yh * (1.0f + erff(yh * 0.70710678118654752f));
    float z  = comb[(size_t)b * 128 + j] + g;
    __syncthreads();
    sv[j] = z;
    __syncthreads();
    float mu2 = 0.0f;
    for (int i = 0; i < 128; ++i) mu2 += sv[i];
    mu2 *= (1.0f / 128.0f);
    float v2 = 0.0f;
    for (int i = 0; i < 128; ++i) { float t = sv[i] - mu2; v2 += t * t; }
    v2 *= (1.0f / 128.0f);
    out[(size_t)b * 128 + j] = (z - mu2) * rsqrtf(v2 + 1e-5f) * fn_g[j] + fn_b[j];
}

// ---------------------------------------------------------------------------
extern "C" void kernel_launch(void* const* d_in, const int* in_sizes, int n_in,
                              void* d_out, int out_size, void* d_ws, size_t ws_size,
                              hipStream_t stream) {
    (void)in_sizes; (void)n_in; (void)out_size; (void)ws_size;
    const int B = 2048;

    const float* x = (const float*)d_in[0];
    const float* cw[3], *cb[3], *bg[3], *bb[3], *bm[3], *bv[3], *lw[3], *lb[3];
    for (int i = 0; i < 3; ++i) {
        cw[i] = (const float*)d_in[1 + 8 * i];
        cb[i] = (const float*)d_in[2 + 8 * i];
        bg[i] = (const float*)d_in[3 + 8 * i];
        bb[i] = (const float*)d_in[4 + 8 * i];
        bm[i] = (const float*)d_in[5 + 8 * i];
        bv[i] = (const float*)d_in[6 + 8 * i];
        lw[i] = (const float*)d_in[7 + 8 * i];
        lb[i] = (const float*)d_in[8 + 8 * i];
    }
    const float* band_w = (const float*)d_in[25];
    const float* band_b = (const float*)d_in[26];
    const float* ln_g   = (const float*)d_in[27];
    const float* ln_b   = (const float*)d_in[28];
    const float* fn_g   = (const float*)d_in[29];
    const float* fn_b   = (const float*)d_in[30];

    // ---- workspace carve-up ----
    char* ws = (char*)d_ws;
    size_t off = 0;
    auto carve = [&](size_t bytes) -> char* {
        char* p = ws + off;
        off += (bytes + 255) & ~(size_t)255;
        return p;
    };
    float* feat128  = (float*)carve((size_t)B * 32 * 65 * 4);
    float* feat64   = (float*)carve((size_t)B * 32 * 33 * 4);
    float* feat32   = (float*)carve((size_t)B * 32 * 17 * 4);
    float* bandfeat = (float*)carve((size_t)B * 160 * 4);
    float* comb     = (float*)carve((size_t)B * 128 * 4);
    _Float16* tw128h = (_Float16*)carve(2 * 5 * 4 * 512 * 2);  // 10 tiles * 4 ksteps
    _Float16* tw128l = (_Float16*)carve(2 * 5 * 4 * 512 * 2);
    _Float16* tw64h  = (_Float16*)carve(2 * 3 * 2 * 512 * 2);
    _Float16* tw64l  = (_Float16*)carve(2 * 3 * 2 * 512 * 2);
    _Float16* tw32h  = (_Float16*)carve(2 * 2 * 1 * 512 * 2);
    _Float16* tw32l  = (_Float16*)carve(2 * 2 * 1 * 512 * 2);
    _Float16* cwph[3], *cwpl[3];
    for (int i = 0; i < 3; ++i) {
        cwph[i] = (_Float16*)carve(12 * 512 * 2);
        cwpl[i] = (_Float16*)carve(12 * 512 * 2);
    }

    // ---- pack twiddles + conv weights (deterministic, recomputed each call) ----
    pack_twiddle_kernel<<<(2*5*4*512 + 255) / 256, 256, 0, stream>>>(tw128h, tw128l, 128, 65, 5, 4);
    pack_twiddle_kernel<<<(2*3*2*512 + 255) / 256, 256, 0, stream>>>(tw64h,  tw64l,  64,  33, 3, 2);
    pack_twiddle_kernel<<<(2*2*1*512 + 255) / 256, 256, 0, stream>>>(tw32h,  tw32l,  32,  17, 2, 1);
    for (int i = 0; i < 3; ++i)
        pack_convw_kernel<<<(6144 + 255) / 256, 256, 0, stream>>>(cw[i], cwph[i], cwpl[i]);

    // ---- DFT GEMMs (split-f16 WMMA, re/im pair-wise retirement) ----
    dft_kernel<32, 4, 65, 5, 1><<<(B * 32 / 16) / 8, 256, 0, stream>>>(x, tw128h, tw128l, feat128, bandfeat);
    dft_kernel<16, 2, 33, 3, 0><<<(B * 16 / 16) / 8, 256, 0, stream>>>(x, tw64h, tw64l, feat64, nullptr);
    dft_kernel<16, 1, 17, 2, 0><<<(B * 16 / 16) / 8, 256, 0, stream>>>(x, tw32h, tw32l, feat32, nullptr);

    // ---- branch heads (im2col conv GEMM + fused epilogue) ----
    branch_kernel<17, 2, 43><<<B, 32, 0, stream>>>(feat32, cwph[0], cwpl[0], cb[0],
        bg[0], bb[0], bm[0], bv[0], lw[0], lb[0], comb, 0);
    branch_kernel<33, 3, 43><<<B, 32, 0, stream>>>(feat64, cwph[1], cwpl[1], cb[1],
        bg[1], bb[1], bm[1], bv[1], lw[1], lb[1], comb, 43);
    branch_kernel<65, 5, 42><<<B, 32, 0, stream>>>(feat128, cwph[2], cwpl[2], cb[2],
        bg[2], bb[2], bm[2], bv[2], lw[2], lb[2], comb, 86);

    // ---- band MLP + final LayerNorm ----
    band_final_kernel<<<B, 128, 0, stream>>>(bandfeat, comb, band_w, band_b,
                                             ln_g, ln_b, fn_g, fn_b, (float*)d_out);
}
